// DC_FeatureAlign_25013889531921
// MI455X (gfx1250) — matrix operand order repaired and checked by reading, not compiled, not run on hardware
//
#include <hip/hip_runtime.h>
#include <hip/hip_bf16.h>
#include <stdint.h>

// ---------------------------------------------------------------------------
// DC_FeatureAlign for MI455X (gfx1250, wave32, WMMA).
// All convolutions are lowered to bf16 WMMA GEMMs (f32 accumulate) over
// pixel-major column buffers; gather/gating/BN are VALU kernels.
// GEMM: B tile double-buffered in LDS (async-to-LDS when available),
// A fragments software-pipelined from global.
// ---------------------------------------------------------------------------

typedef __bf16 bf16_t;
typedef __attribute__((ext_vector_type(16))) __bf16 v16bf;
typedef __attribute__((ext_vector_type(8)))  __bf16 v8bf;
typedef __attribute__((ext_vector_type(8)))  float   v8f;
typedef __attribute__((ext_vector_type(4)))  int     v4i;

#define B_    4
#define C_    128
#define CT_   64
#define H_    64
#define W_    64
#define HW_   4096
#define M_    16384          // B*H*W pixels
#define K3_   1152           // 128 * 9
#define NPIX_MASK 16383
#define LDSROW 40            // bf16 elems per LDS row (32 data + 8 pad -> conflict-free)

#if __has_builtin(__builtin_amdgcn_global_load_async_to_lds_b128) && \
    __has_builtin(__builtin_amdgcn_s_wait_asynccnt)
#define HAS_ASYNC_LDS 1
#else
#define HAS_ASYNC_LDS 0
#endif

// ---------------------------------------------------------------------------
// WMMA GEMM:  C[M][NT*16] = A[M][K] * B[NT*16][K]^T + bias
// A row-major bf16, Bw row-major bf16 (NT*16 rows of length K), C f32.
// One wave -> 16 rows x NT*16 cols; 4 waves/block -> 64 rows/block.
// Requires M % 64 == 0, K % 32 == 0.
// ---------------------------------------------------------------------------
template <int NT>
__global__ __launch_bounds__(128)
void gemm_wmma_bf16(const bf16_t* __restrict__ A,
                    const bf16_t* __restrict__ Bw,
                    const float*  __restrict__ bias,
                    float*        __restrict__ C,
                    int M, int K) {
    constexpr int NT16 = NT * 16;
    __shared__ bf16_t lds_b[2][NT16 * LDSROW];

    const int tid   = threadIdx.x;
    const int lane  = tid & 31;
    const int wave  = tid >> 5;
    const int row0  = (blockIdx.x * 4 + wave) * 16;      // wave-uniform
    const int lhalf = lane >> 4;                         // 0/1
    const int lm    = lane & 15;

    // ---- B tile stager: thread t copies row n = t (64B) when t < NT16.
    // (tid < NT16) is wave-uniform: NT16 is a multiple of 32.
    const bool stager = (tid < NT16);
    const bf16_t* bsrc = Bw + (size_t)(stager ? tid : 0) * K;

#if HAS_ASYNC_LDS
    // async global->LDS copy of one k-step tile into buffer `buf`
    auto stage_async = [&](int kb, int buf) {
        if (stager) {
            bf16_t* drow = &lds_b[buf][tid * LDSROW];
            #pragma unroll
            for (int c = 0; c < 4; ++c) {
                __builtin_amdgcn_global_load_async_to_lds_b128(
                    (v4i*)(bsrc + kb + c * 8), (v4i*)(drow + c * 8), 0, 0);
            }
        }
    };
    auto stage_wait = [&]() {
        if (stager) __builtin_amdgcn_s_wait_asynccnt(0);  // wave-uniform branch
        __syncthreads();
    };
    stage_async(0, 0);
    stage_wait();
#else
    v16bf st0, st1;
    auto stage_fetch = [&](int kb) {
        if (stager) {
            st0 = *(const v16bf*)(bsrc + kb);
            st1 = *(const v16bf*)(bsrc + kb + 16);
        }
    };
    auto stage_commit = [&](int buf) {
        if (stager) {
            bf16_t* drow = &lds_b[buf][tid * LDSROW];
            *(v16bf*)(drow)      = st0;
            *(v16bf*)(drow + 16) = st1;
        }
        __syncthreads();
    };
    stage_fetch(0);
    stage_commit(0);
#endif

    v8f acc[NT];
    #pragma unroll
    for (int t = 0; t < NT; ++t)
        #pragma unroll
        for (int i = 0; i < 8; ++i) acc[t][i] = 0.0f;

    const bf16_t* arow = A + (size_t)(row0 + lm) * K + 16 * lhalf;

    // A fragment pipeline registers (16-bit A 16x32 layout: two 16B runs)
    v8bf alo = *(const v8bf*)(arow);
    v8bf ahi = *(const v8bf*)(arow + 8);

    for (int kb = 0; kb < K; kb += 32) {
        const int  cur  = (kb >> 5) & 1;
        const bool more = (kb + 32) < K;

        // ---- prefetch next tile (global traffic overlaps wmma chain) ----
#if HAS_ASYNC_LDS
        if (more) stage_async(kb + 32, cur ^ 1);
#else
        if (more) stage_fetch(kb + 32);
#endif
        v8bf nalo, nahi;
        if (more) {
            nalo = *(const v8bf*)(arow + kb + 32);
            nahi = *(const v8bf*)(arow + kb + 40);
        }

        v16bf a;
        #pragma unroll
        for (int i = 0; i < 8; ++i) { a[i] = alo[i]; a[8 + i] = ahi[i]; }

        // ---- 8 (or 2) WMMAs fed from LDS ----
        const bf16_t* bufc = lds_b[cur];
        #pragma unroll
        for (int t = 0; t < NT; ++t) {
            const bf16_t* brow = bufc + (t * 16 + lm) * LDSROW + 16 * lhalf;
            v16bf b = *(const v16bf*)brow;   // ds_load_b128 x2, conflict-free
            acc[t] = __builtin_amdgcn_wmma_f32_16x16x32_bf16(
                         false, a, false, b, (short)0, acc[t], false, false);
        }

        alo = nalo; ahi = nahi;

        __syncthreads();                      // everyone done reading buf `cur`
#if HAS_ASYNC_LDS
        if (more) stage_wait();
#else
        if (more) stage_commit(cur ^ 1);
#endif
    }

    const int Npad = NT16;
    #pragma unroll
    for (int t = 0; t < NT; ++t) {
        const int n  = t * 16 + lm;
        const float bv = bias ? bias[n] : 0.0f;
        #pragma unroll
        for (int r = 0; r < 8; ++r) {
            const int m = row0 + r + 8 * lhalf;          // 32-bit C/D layout
            C[(size_t)m * Npad + n] = acc[t][r] + bv;
        }
    }
}

// ---------------------------------------------------------------------------
// Pack feature_top (NCHW f32) -> pixel-major bf16 column [p][ci]
// ---------------------------------------------------------------------------
__global__ void pack_top_col(const float* __restrict__ ftop,
                             bf16_t* __restrict__ colTop) {
    int tid = blockIdx.x * blockDim.x + threadIdx.x;     // B*CT*HW
    if (tid >= B_ * CT_ * HW_) return;
    int hw = tid & (HW_ - 1);
    int ci = (tid >> 12) & (CT_ - 1);
    int b  = tid >> 18;
    colTop[((size_t)(b * HW_ + hw)) * CT_ + ci] = (bf16_t)ftop[tid];
}

// ---------------------------------------------------------------------------
// Convert weights to bf16 GEMM form; build padded offset/mask bias.
// wB_off rows: 0..17 = p_w, 18..26 = m_w, 27..31 = 0    (k = ci*9 + tap)
// ---------------------------------------------------------------------------
__global__ void prep_weights(const float* __restrict__ w_exp,
                             const float* __restrict__ p_w,
                             const float* __restrict__ m_w,
                             const float* __restrict__ conv_w,
                             const float* __restrict__ p_b,
                             const float* __restrict__ m_b,
                             bf16_t* __restrict__ wB_exp,
                             bf16_t* __restrict__ wB_off,
                             bf16_t* __restrict__ wB_conv,
                             float*  __restrict__ bias_off) {
    int tid = blockIdx.x * blockDim.x + threadIdx.x;
    if (tid < C_ * CT_)  wB_exp[tid]  = (bf16_t)w_exp[tid];
    if (tid < 32 * K3_) {
        int r = tid / K3_, k = tid - r * K3_;
        float v = (r < 18) ? p_w[r * K3_ + k]
                           : ((r < 27) ? m_w[(r - 18) * K3_ + k] : 0.0f);
        wB_off[tid] = (bf16_t)v;
    }
    if (tid < C_ * K3_) wB_conv[tid] = (bf16_t)conv_w[tid];
    if (tid < 32) bias_off[tid] = (tid < 18) ? p_b[tid]
                                 : ((tid < 27) ? m_b[tid - 18] : 0.0f);
}

// ---------------------------------------------------------------------------
// AFM-Lite gating: v0/v1 1x1 convs -> lw -> softmax -> fused blend.
// ---------------------------------------------------------------------------
__global__ void gating_fuse(const float* __restrict__ bottom,
                            const float* __restrict__ top_pm,
                            const float* __restrict__ w_l0, const float* __restrict__ b_l0,
                            const float* __restrict__ w_l1, const float* __restrict__ b_l1,
                            const float* __restrict__ w_lv, const float* __restrict__ b_lv,
                            float* __restrict__ fused_pm) {
    int p = blockIdx.x * blockDim.x + threadIdx.x;
    if (p >= M_) return;
    int b = p >> 12, hw = p & (HW_ - 1);
    const float* xb = bottom + ((size_t)b * C_) * HW_ + hw;  // stride HW_ per ci
    const float* xt = top_pm + (size_t)p * C_;

    float v0[16], v1[16];
    #pragma unroll
    for (int c = 0; c < 16; ++c) { v0[c] = b_l0[c]; v1[c] = b_l1[c]; }
    for (int ci = 0; ci < C_; ++ci) {
        float vb = xb[(size_t)ci * HW_];
        float vt = xt[ci];
        #pragma unroll
        for (int c = 0; c < 16; ++c) {
            v0[c] = fmaf(vb, w_l0[c * C_ + ci], v0[c]);
            v1[c] = fmaf(vt, w_l1[c * C_ + ci], v1[c]);
        }
    }
    float l0 = b_lv[0], l1 = b_lv[1];
    #pragma unroll
    for (int c = 0; c < 16; ++c) {
        l0 += v0[c] * w_lv[c]      + v1[c] * w_lv[16 + c];
        l1 += v0[c] * w_lv[32 + c] + v1[c] * w_lv[48 + c];
    }
    float mx = fmaxf(l0, l1);
    float e0 = __expf(l0 - mx), e1 = __expf(l1 - mx);
    float inv = 1.0f / (e0 + e1);
    float s0 = e0 * inv, s1 = e1 * inv;

    float* fo = fused_pm + (size_t)p * C_;
    for (int ci = 0; ci < C_; ++ci)
        fo[ci] = xb[(size_t)ci * HW_] * s0 + xt[ci] * s1;
}

// ---------------------------------------------------------------------------
// im2col (3x3, pad 1) of fused_pm -> bf16 column [p][ci*9+tap]
// ---------------------------------------------------------------------------
__global__ void pack_fused_col(const float* __restrict__ fused_pm,
                               bf16_t* __restrict__ col) {
    int tid = blockIdx.x * blockDim.x + threadIdx.x;     // M_*C_
    if (tid >= M_ * C_) return;
    int ci = tid & (C_ - 1);
    int p  = tid >> 7;
    int b = p >> 12, h = (p >> 6) & 63, w = p & 63;
    bf16_t* out = col + (size_t)p * K3_ + ci * 9;
    #pragma unroll
    for (int tap = 0; tap < 9; ++tap) {
        int hh = h + tap / 3 - 1, ww = w + tap % 3 - 1;
        float v = 0.0f;
        if (hh >= 0 && hh < H_ && ww >= 0 && ww < W_)
            v = fused_pm[(((size_t)b * HW_) + hh * W_ + ww) * C_ + ci];
        out[tap] = (bf16_t)v;
    }
}

// ---------------------------------------------------------------------------
// Deformable bilinear sampling + modulation -> bf16 column [p][ci*9+tap]
// offmask rows: [0..8]=off_x, [9..17]=off_y, [18..26]=mask logits
// ---------------------------------------------------------------------------
__global__ void deform_gather(const float* __restrict__ fused_pm,
                              const float* __restrict__ offmask,
                              bf16_t* __restrict__ col) {
    int tid = blockIdx.x * blockDim.x + threadIdx.x;     // M_*9
    if (tid >= M_ * 9) return;
    int tap = tid % 9;
    int p   = tid / 9;
    int b = p >> 12, h = (p >> 6) & 63, w = p & 63;

    const float* om = offmask + (size_t)p * 32;
    float offx = om[tap], offy = om[9 + tap];
    float mm   = 1.0f / (1.0f + __expf(-om[18 + tap]));

    float px = (float)(h + 1) + (float)(tap / 3 - 1) + offx;
    float py = (float)(w + 1) + (float)(tap % 3 - 1) + offy;
    float qltx = floorf(px), qlty = floorf(py);
    float qrbx = qltx + 1.0f, qrby = qlty + 1.0f;
    qltx = fminf(fmaxf(qltx, 0.0f), 65.0f);
    qlty = fminf(fmaxf(qlty, 0.0f), 65.0f);
    qrbx = fminf(fmaxf(qrbx, 0.0f), 65.0f);
    qrby = fminf(fmaxf(qrby, 0.0f), 65.0f);
    px   = fminf(fmaxf(px,   0.0f), 65.0f);
    py   = fminf(fmaxf(py,   0.0f), 65.0f);

    float g_lt = (1.0f + qltx - px) * (1.0f + qlty - py);
    float g_rb = (1.0f - qrbx + px) * (1.0f - qrby + py);
    float g_lb = (1.0f + qltx - px) * (1.0f - qrby + py);
    float g_rt = (1.0f - qrbx + px) * (1.0f + qlty - py);

    const float* base = fused_pm + (size_t)b * HW_ * C_;
    int ixl = (int)qltx, iyl = (int)qlty, ixr = (int)qrbx, iyr = (int)qrby;
    // padded array has zeros on the 1-pixel border: valid interior is [1,64]
    const float* plt = (ixl >= 1 && ixl <= 64 && iyl >= 1 && iyl <= 64)
                       ? base + ((size_t)(ixl - 1) * W_ + (iyl - 1)) * C_ : nullptr;
    const float* prb = (ixr >= 1 && ixr <= 64 && iyr >= 1 && iyr <= 64)
                       ? base + ((size_t)(ixr - 1) * W_ + (iyr - 1)) * C_ : nullptr;
    const float* plb = (ixl >= 1 && ixl <= 64 && iyr >= 1 && iyr <= 64)
                       ? base + ((size_t)(ixl - 1) * W_ + (iyr - 1)) * C_ : nullptr;
    const float* prt = (ixr >= 1 && ixr <= 64 && iyl >= 1 && iyl <= 64)
                       ? base + ((size_t)(ixr - 1) * W_ + (iyl - 1)) * C_ : nullptr;

    bf16_t* out = col + (size_t)p * K3_ + tap;
    for (int ci = 0; ci < C_; ++ci) {
        float v = 0.0f;
        if (plt) v = fmaf(g_lt, plt[ci], v);
        if (prb) v = fmaf(g_rb, prb[ci], v);
        if (plb) v = fmaf(g_lb, plb[ci], v);
        if (prt) v = fmaf(g_rt, prt[ci], v);
        out[(size_t)ci * 9] = (bf16_t)(v * mm);
    }
}

// ---------------------------------------------------------------------------
// BN statistics: per-channel sum / sum-of-squares over all pixels.
// ---------------------------------------------------------------------------
__global__ void bn_reduce(const float* __restrict__ dc,
                          float* __restrict__ sums,
                          float* __restrict__ sumsq) {
    int tid = blockIdx.x * blockDim.x + threadIdx.x;     // 65536 threads
    int co = tid & (C_ - 1);
    int ps = tid >> 7;                                   // 512 slices per channel
    float s = 0.0f, s2 = 0.0f;
    for (int p = ps; p < M_; p += 512) {
        float v = dc[(size_t)p * C_ + co];
        s += v; s2 = fmaf(v, v, s2);
    }
    atomicAdd(&sums[co], s);
    atomicAdd(&sumsq[co], s2);
}

// ---------------------------------------------------------------------------
// BN + ReLU + residual, pixel-major -> NCHW
// ---------------------------------------------------------------------------
__global__ void bn_relu_res(const float* __restrict__ dc,
                            const float* __restrict__ sums,
                            const float* __restrict__ sumsq,
                            const float* __restrict__ gamma,
                            const float* __restrict__ beta,
                            const float* __restrict__ bottom,
                            float* __restrict__ out) {
    int tid = blockIdx.x * blockDim.x + threadIdx.x;     // C_*M_
    if (tid >= C_ * M_) return;
    int p  = tid & NPIX_MASK;
    int co = tid >> 14;
    int b = p >> 12, hw = p & (HW_ - 1);

    const float rM = 1.0f / (float)M_;
    float mean = sums[co] * rM;
    float var  = sumsq[co] * rM - mean * mean;
    float inv  = rsqrtf(var + 1e-5f);
    float v = (dc[(size_t)p * C_ + co] - mean) * inv * gamma[co] + beta[co];
    v = fmaxf(v, 0.0f);
    size_t oi = (((size_t)b * C_) + co) * HW_ + hw;
    out[oi] = v + bottom[oi];
}

// ---------------------------------------------------------------------------
extern "C" void kernel_launch(void* const* d_in, const int* in_sizes, int n_in,
                              void* d_out, int out_size, void* d_ws, size_t ws_size,
                              hipStream_t stream) {
    (void)in_sizes; (void)n_in; (void)out_size; (void)ws_size;
    const float* bottom = (const float*)d_in[0];
    const float* ftop   = (const float*)d_in[1];
    const float* w_l0   = (const float*)d_in[2];
    const float* b_l0   = (const float*)d_in[3];
    const float* w_l1   = (const float*)d_in[4];
    const float* b_l1   = (const float*)d_in[5];
    const float* w_lv   = (const float*)d_in[6];
    const float* b_lv   = (const float*)d_in[7];
    const float* w_exp  = (const float*)d_in[8];
    const float* b_exp  = (const float*)d_in[9];
    const float* p_w    = (const float*)d_in[10];
    const float* p_b    = (const float*)d_in[11];
    const float* m_w    = (const float*)d_in[12];
    const float* m_b    = (const float*)d_in[13];
    const float* conv_w = (const float*)d_in[14];
    const float* gamma  = (const float*)d_in[15];
    const float* beta   = (const float*)d_in[16];
    float* out = (float*)d_out;

    // ---- workspace carving (256B aligned) ----
    uint8_t* ws = (uint8_t*)d_ws;
    size_t off = 0;
    auto carve = [&](size_t bytes) -> void* {
        void* ptr = (void*)(ws + off);
        off = (off + bytes + 255) & ~(size_t)255;
        return ptr;
    };
    float*  top_pm     = (float*)carve((size_t)M_ * C_ * 4);     // 8 MB
    float*  fused_pm   = (float*)carve((size_t)M_ * C_ * 4);     // 8 MB
    float*  offmask_pm = (float*)carve((size_t)M_ * 32 * 4);     // 2 MB
    float*  dc_pm      = (float*)carve((size_t)M_ * C_ * 4);     // 8 MB
    float*  sums       = (float*)carve(256 * 4);                 // sums[128] + sumsq[128]
    float*  sumsq      = sums + 128;
    float*  bias_off   = (float*)carve(32 * 4);
    bf16_t* colTop     = (bf16_t*)carve((size_t)M_ * CT_ * 2);   // 2 MB
    bf16_t* col_fused  = (bf16_t*)carve((size_t)M_ * K3_ * 2);   // 37.75 MB
    bf16_t* col_def    = (bf16_t*)carve((size_t)M_ * K3_ * 2);   // 37.75 MB
    bf16_t* wB_exp     = (bf16_t*)carve((size_t)C_ * CT_ * 2);
    bf16_t* wB_off     = (bf16_t*)carve((size_t)32 * K3_ * 2);
    bf16_t* wB_conv    = (bf16_t*)carve((size_t)C_ * K3_ * 2);

    // ---- 0. weights -> bf16 GEMM form ----
    prep_weights<<<(C_ * K3_ + 255) / 256, 256, 0, stream>>>(
        w_exp, p_w, m_w, conv_w, p_b, m_b, wB_exp, wB_off, wB_conv, bias_off);

    // ---- 1. pack top feature, expand channels via WMMA GEMM ----
    pack_top_col<<<(B_ * CT_ * HW_ + 255) / 256, 256, 0, stream>>>(ftop, colTop);
    gemm_wmma_bf16<8><<<M_ / 64, 128, 0, stream>>>(colTop, wB_exp, b_exp, top_pm, M_, CT_);

    // ---- 2. AFM gating + blend ----
    gating_fuse<<<(M_ + 127) / 128, 128, 0, stream>>>(
        bottom, top_pm, w_l0, b_l0, w_l1, b_l1, w_lv, b_lv, fused_pm);

    // ---- 3. offset + mask 3x3 conv via WMMA GEMM (fused, N padded to 32) ----
    pack_fused_col<<<(M_ * C_ + 255) / 256, 256, 0, stream>>>(fused_pm, col_fused);
    gemm_wmma_bf16<2><<<M_ / 64, 128, 0, stream>>>(col_fused, wB_off, bias_off, offmask_pm, M_, K3_);

    // ---- 4. deformable bilinear sampling + modulation ----
    deform_gather<<<(M_ * 9 + 255) / 256, 256, 0, stream>>>(fused_pm, offmask_pm, col_def);

    // ---- 5. big deform conv via WMMA GEMM ----
    gemm_wmma_bf16<8><<<M_ / 64, 128, 0, stream>>>(col_def, wB_conv, nullptr, dc_pm, M_, K3_);

    // ---- 6. BN stats + BN/ReLU/residual epilogue ----
    (void)hipMemsetAsync(sums, 0, 256 * sizeof(float), stream);
    bn_reduce<<<256, 256, 0, stream>>>(dc_pm, sums, sumsq);
    bn_relu_res<<<(C_ * M_ + 255) / 256, 256, 0, stream>>>(
        dc_pm, sums, sumsq, gamma, beta, bottom, out);
}
